// ProteinVAEVector_49581102465461
// MI455X (gfx1250) — matrix operand order repaired and checked by reading, not compiled, not run on hardware
//
#include <hip/hip_runtime.h>
#include <hip/hip_bf16.h>

typedef __attribute__((ext_vector_type(16))) _Float16 v16h;
typedef __attribute__((ext_vector_type(8)))  _Float16 v8h;
typedef __attribute__((ext_vector_type(8)))  float    v8f;
typedef __attribute__((ext_vector_type(4)))  float    v4f;

#define WAVES 4
#define H1_STRIDE 264   // halves per row (256 + 8 pad), 528B, mult of 16
#define H2_STRIDE 68    // floats per row (64 + 4 pad), 272B, mult of 16

// ---------------------------------------------------------------------------
// Weight pack: f32 [K][N] row-major  ->  f16 WMMA B-fragment order so each
// lane's fragment (16 halves) is one contiguous 32B chunk.
// B-frag layout (16-bit B, 32x16): lane l holds N = l&15,
//   half h (0..15) holds K = kt*32 + ((l>>4)<<3) + (h&7) + ((h>>3)<<4).
// ---------------------------------------------------------------------------
__global__ void pack_w_kernel(const float* __restrict__ W, _Float16* __restrict__ dst,
                              int K, int N) {
  int idx = blockIdx.x * blockDim.x + threadIdx.x;
  if (idx >= K * N) return;
  int k = idx / N, n = idx % N;
  int kt = k >> 5, kk = k & 31;
  int lane = (((kk >> 3) & 1) << 4) | (n & 15);
  int h = (kk & 7) | ((kk >> 4) << 3);
  int nt = n >> 4;
  int numNt = N >> 4;
  size_t d = ((((size_t)kt * numNt + nt) * 32 + lane) << 4) + h;
  dst[d] = (_Float16)W[idx];
}

// ---------------------------------------------------------------------------
// Fragment helpers
// ---------------------------------------------------------------------------
static __device__ __forceinline__ v8f wmma16(v16h a, v16h b, v8f c) {
  return __builtin_amdgcn_wmma_f32_16x16x32_f16(false, a, false, b, (short)0, c,
                                                false, false);
}

// B fragment: contiguous 32B per lane from packed weights.
static __device__ __forceinline__ v16h ldB(const _Float16* __restrict__ pw,
                                           int numNt, int kt, int nt, int lane) {
  return *(const v16h*)(pw + ((((size_t)kt * numNt + nt) * 32 + lane) << 4));
}

// A fragment from global f32 (16-bit A layout: per-lane chunks K={0..7} and
// K={16..23} relative to k0 = kt*32 + 8*(lane>=16)). p0 is 16B aligned.
static __device__ __forceinline__ v16h ldA_g(const float* __restrict__ p0) {
  const v4f* q = (const v4f*)p0;
  v4f c0 = q[0], c1 = q[1], c2 = q[4], c3 = q[5];
  v16h a;
#pragma unroll
  for (int i = 0; i < 4; ++i) {
    a[i]      = (_Float16)c0[i];
    a[i + 4]  = (_Float16)c1[i];
    a[i + 8]  = (_Float16)c2[i];
    a[i + 12] = (_Float16)c3[i];
  }
  return a;
}

// A fragment from LDS f16 (same chunking; base 16B aligned).
static __device__ __forceinline__ v16h ldA_s(const _Float16* base) {
  const v8h* p = (const v8h*)base;
  v8h c0 = p[0];   // K k0..k0+7
  v8h c1 = p[2];   // K k0+16..k0+23
  v16h a;
#pragma unroll
  for (int i = 0; i < 8; ++i) { a[i] = c0[i]; a[i + 8] = c1[i]; }
  return a;
}

static __device__ __forceinline__ v8f splat8(float v) {
  v8f c;
#pragma unroll
  for (int i = 0; i < 8; ++i) c[i] = v;
  return c;
}

#define LDS_FENCE() asm volatile("s_wait_dscnt 0" ::: "memory")

// ---------------------------------------------------------------------------
// Fused VAE forward. One wave = 16 rows through the whole network.
// ---------------------------------------------------------------------------
__global__ __launch_bounds__(WAVES * 32) void vae_fused_kernel(
    const float* __restrict__ x, const float* __restrict__ eps,
    const _Float16* __restrict__ pW1, const float* __restrict__ b1,
    const _Float16* __restrict__ pW2, const float* __restrict__ b2,
    const float* __restrict__ Wx1, const float* __restrict__ bx1,
    const float* __restrict__ Wx2, const float* __restrict__ bx2,
    const float* __restrict__ Ws1, const float* __restrict__ bs1,
    const float* __restrict__ Ws2, const float* __restrict__ bs2,
    const _Float16* __restrict__ pWd1, const float* __restrict__ bd1,
    const _Float16* __restrict__ pWd2, const float* __restrict__ bd2,
    float* __restrict__ out, int Btot) {
  __shared__ _Float16 s_act[WAVES][16 * H1_STRIDE];  // h1, reused for dec-L1 out
  __shared__ float    s_h2[WAVES][16 * H2_STRIDE];
  __shared__ float    s_mu[WAVES][16];
  __shared__ float    s_sg[WAVES][16];
  __shared__ float    s_hw[2][64 * 16];  // head layer1 weights (mu, logsig)
  __shared__ float    s_hb[2][16];
  __shared__ float    s_hw2[2][16];
  __shared__ float    s_hb2[2];

  const int tid = threadIdx.x;
  // Cooperative preload of head weights (shared by all waves).
  for (int i = tid; i < 1024; i += WAVES * 32) {
    s_hw[0][i] = Wx1[i];
    s_hw[1][i] = Ws1[i];
  }
  if (tid < 16) {
    s_hb[0][tid]  = bx1[tid];  s_hb[1][tid]  = bs1[tid];
    s_hw2[0][tid] = Wx2[tid];  s_hw2[1][tid] = Ws2[tid];
  }
  if (tid < 2) s_hb2[tid] = (tid == 0) ? bx2[0] : bs2[0];
  __syncthreads();

  const int lane = tid & 31;
  const int wid  = tid >> 5;
  const int row0 = (blockIdx.x * WAVES + wid) << 4;
  const int rlo  = lane & 15;        // column within tile for C/B, M for A
  const int hb8  = (lane >> 4) << 3; // A-layout K chunk offset / C row offset
  const size_t MU_OFF = (size_t)Btot * 768;
  const size_t LS_OFF = MU_OFF + (size_t)Btot;

  _Float16* act = s_act[wid];
  float*    h2  = s_h2[wid];

  // ---------------- Encoder L1: [16,768] x [768,256] + b1, ReLU -----------
  v8f acc[16];
#pragma unroll
  for (int nt = 0; nt < 16; ++nt) acc[nt] = splat8(b1[nt * 16 + rlo]);
  for (int kt = 0; kt < 24; ++kt) {
    const float* px = x + (size_t)(row0 + rlo) * 768 + kt * 32 + hb8;
    v16h a = ldA_g(px);
#pragma unroll
    for (int nt = 0; nt < 16; ++nt)
      acc[nt] = wmma16(a, ldB(pW1, 16, kt, nt, lane), acc[nt]);
  }
#pragma unroll
  for (int nt = 0; nt < 16; ++nt)
#pragma unroll
    for (int v = 0; v < 8; ++v)
      act[(v + hb8) * H1_STRIDE + nt * 16 + rlo] =
          (_Float16)fmaxf(acc[nt][v], 0.f);
  LDS_FENCE();

  // ---------------- Encoder L2: [16,256] x [256,64] + b2, ReLU ------------
  v8f a2[4];
#pragma unroll
  for (int nt = 0; nt < 4; ++nt) a2[nt] = splat8(b2[nt * 16 + rlo]);
  for (int kt = 0; kt < 8; ++kt) {
    v16h a = ldA_s(act + rlo * H1_STRIDE + kt * 32 + hb8);
#pragma unroll
    for (int nt = 0; nt < 4; ++nt)
      a2[nt] = wmma16(a, ldB(pW2, 4, kt, nt, lane), a2[nt]);
  }
#pragma unroll
  for (int nt = 0; nt < 4; ++nt)
#pragma unroll
    for (int v = 0; v < 8; ++v)
      h2[(v + hb8) * H2_STRIDE + nt * 16 + rlo] = fmaxf(a2[nt][v], 0.f);
  LDS_FENCE();

  // ---------------- Heads: lanes 0-15 -> mu, lanes 16-31 -> logsigma ------
  float hv[64];
  {
    const v4f* hp = (const v4f*)(h2 + rlo * H2_STRIDE);
#pragma unroll
    for (int c = 0; c < 16; ++c) {
      v4f t = hp[c];
#pragma unroll
      for (int i = 0; i < 4; ++i) hv[c * 4 + i] = t[i];
    }
  }
  const int side = lane >> 4;
  const float* W1h = s_hw[side];
  const float* bbh = s_hb[side];
  const float* W2h = s_hw2[side];
  float val = s_hb2[side];
  for (int j = 0; j < 16; ++j) {
    float s = bbh[j];
#pragma unroll
    for (int k = 0; k < 64; ++k) s = fmaf(hv[k], W1h[(k << 4) + j], s);
    val = fmaf(fmaxf(s, 0.f), W2h[j], val);
  }
  if (side == 0) {
    out[MU_OFF + row0 + rlo] = val;
    s_mu[wid][rlo] = val;
  } else {
    out[LS_OFF + row0 + rlo] = val;
    s_sg[wid][rlo] = expf(val);
  }
  LDS_FENCE();

  // ---------------- Reparam: z = mu*(1+sigma) + sigma^2 * eps -------------
  const float mu = s_mu[wid][rlo];
  const float sg = s_sg[wid][rlo];
  const float m1 = fmaf(mu, sg, mu);
  const float s2 = sg * sg;
  v16h zf[4];
#pragma unroll
  for (int kt = 0; kt < 4; ++kt) {
    const v4f* pe =
        (const v4f*)(eps + (size_t)(row0 + rlo) * 128 + kt * 32 + hb8);
    v4f c0 = pe[0], c1 = pe[1], c2 = pe[4], c3 = pe[5];
#pragma unroll
    for (int i = 0; i < 4; ++i) {
      zf[kt][i]      = (_Float16)fmaf(s2, c0[i], m1);
      zf[kt][i + 4]  = (_Float16)fmaf(s2, c1[i], m1);
      zf[kt][i + 8]  = (_Float16)fmaf(s2, c2[i], m1);
      zf[kt][i + 12] = (_Float16)fmaf(s2, c3[i], m1);
    }
  }

  // ---------------- Decoder L1: [16,128] x [128,256] + bd1, ReLU ----------
  for (int nt = 0; nt < 16; ++nt) {
    v8f c = splat8(bd1[nt * 16 + rlo]);
#pragma unroll
    for (int kt = 0; kt < 4; ++kt)
      c = wmma16(zf[kt], ldB(pWd1, 16, kt, nt, lane), c);
#pragma unroll
    for (int v = 0; v < 8; ++v)
      act[(v + hb8) * H1_STRIDE + nt * 16 + rlo] = (_Float16)fmaxf(c[v], 0.f);
  }
  LDS_FENCE();

  // ---------------- Decoder L2: [16,256] x [256,768] + bd2 -> out ---------
  v16h afr[8];
#pragma unroll
  for (int kt = 0; kt < 8; ++kt)
    afr[kt] = ldA_s(act + rlo * H1_STRIDE + kt * 32 + hb8);
  for (int nt = 0; nt < 48; ++nt) {
    v8f c = splat8(bd2[nt * 16 + rlo]);
#pragma unroll
    for (int kt = 0; kt < 8; ++kt)
      c = wmma16(afr[kt], ldB(pWd2, 48, kt, nt, lane), c);
#pragma unroll
    for (int v = 0; v < 8; ++v)
      out[(size_t)(row0 + v + hb8) * 768 + nt * 16 + rlo] = c[v];
  }
}

// ---------------------------------------------------------------------------
extern "C" void kernel_launch(void* const* d_in, const int* in_sizes, int n_in,
                              void* d_out, int out_size, void* d_ws,
                              size_t ws_size, hipStream_t stream) {
  const float* x   = (const float*)d_in[0];
  const float* eps = (const float*)d_in[1];
  const float* W1  = (const float*)d_in[2];
  const float* b1  = (const float*)d_in[3];
  const float* W2  = (const float*)d_in[4];
  const float* b2  = (const float*)d_in[5];
  const float* Wx1 = (const float*)d_in[6];
  const float* bx1 = (const float*)d_in[7];
  const float* Wx2 = (const float*)d_in[8];
  const float* bx2 = (const float*)d_in[9];
  const float* Ws1 = (const float*)d_in[10];
  const float* bs1 = (const float*)d_in[11];
  const float* Ws2 = (const float*)d_in[12];
  const float* bs2 = (const float*)d_in[13];
  const float* Wd1 = (const float*)d_in[14];
  const float* bd1 = (const float*)d_in[15];
  const float* Wd2 = (const float*)d_in[16];
  const float* bd2 = (const float*)d_in[17];
  float* out = (float*)d_out;

  const int Btot = in_sizes[0] / 768;

  // f16 packed weights in workspace
  _Float16* ws   = (_Float16*)d_ws;
  _Float16* pW1  = ws;
  _Float16* pW2  = pW1 + 768 * 256;
  _Float16* pWd1 = pW2 + 256 * 64;
  _Float16* pWd2 = pWd1 + 128 * 256;

  auto pack = [&](const float* W, _Float16* dst, int K, int N) {
    int n = K * N;
    pack_w_kernel<<<(n + 255) / 256, 256, 0, stream>>>(W, dst, K, N);
  };
  pack(W1, pW1, 768, 256);
  pack(W2, pW2, 256, 64);
  pack(Wd1, pWd1, 128, 256);
  pack(Wd2, pWd2, 256, 768);

  const int blocks = Btot / (16 * WAVES);
  vae_fused_kernel<<<blocks, WAVES * 32, 0, stream>>>(
      x, eps, pW1, b1, pW2, b2, Wx1, bx1, Wx2, bx2, Ws1, bs1, Ws2, bs2, pWd1,
      bd1, pWd2, bd2, out, Btot);
}
